// WindowAttention3D_18176301596785
// MI455X (gfx1250) — compile-verified, matrix operand order checked by reading, not run on hardware
//
#include <hip/hip_runtime.h>

// ---------------------------------------------------------------------------
// Swin-3D window attention for MI455X (gfx1250, wave32, WMMA).
// bias/mask pad+fold -> QKV GEMM (bf16 wmma, double-buffered LDS) ->
// transposed flash attention (S^T = K*Q^T, O^T = V^T*P^T: lane-local softmax,
// zero-shuffle P reuse, raw v_exp_f32) -> proj GEMM (bf16 wmma, f32 out).
// ---------------------------------------------------------------------------

typedef __attribute__((ext_vector_type(16))) __bf16 bf16x16;
typedef __attribute__((ext_vector_type(8)))  __bf16 bf16x8;
typedef __attribute__((ext_vector_type(8)))  float  v8f;
typedef __attribute__((ext_vector_type(4)))  float  f32x4;

#define HEADS 8
#define HD 32
#define NTOK 343
#define NPAD 352            // 22 * 16
#define NT 22               // 16-wide key/query tiles
#define BW 256
#define NW 64
#define CDIM 256
#define LOG2E 1.4426950408889634f

static __device__ __forceinline__ v8f wmma_bf16(const bf16x16& a, const bf16x16& b, const v8f& c) {
  return __builtin_amdgcn_wmma_f32_16x16x32_bf16(false, a, false, b, (short)0, c, false, false);
}

// Fragment loader: base points at this lane's 32-element k-run; kb = 8*(lane>=16).
// Layout: elems 0..7 <-> k = kb..kb+7, elems 8..15 <-> k = kb+16..kb+23.
static __device__ __forceinline__ bf16x16 load_frag(const __bf16* base, int kb) {
  bf16x8 lo = *(const bf16x8*)(base + kb);
  bf16x8 hi = *(const bf16x8*)(base + kb + 16);
  bf16x16 f;
#pragma unroll
  for (int i = 0; i < 8; ++i) { f[i] = lo[i]; f[8 + i] = hi[i]; }
  return f;
}

// ---------------------------------------------------------------------------
// Kernel 1a: biasP[h][q][keyP] = rpb[rel_idx[q][key]][h]*log2e  (pad keys -> 0)
// ---------------------------------------------------------------------------
__global__ __launch_bounds__(256) void bias_pad_k(const float* __restrict__ tbl,
                                                  const int* __restrict__ ridx,
                                                  float* __restrict__ biasP) {
  int e = blockIdx.x * 256 + threadIdx.x;           // over NTOK*NPAD
  if (e >= NTOK * NPAD) return;
  int q = e / NPAD, key = e - q * NPAD;
  if (key < NTOK) {
    int idx = ridx[q * NTOK + key];
#pragma unroll
    for (int h = 0; h < HEADS; ++h)
      biasP[(size_t)h * NTOK * NPAD + e] = tbl[(size_t)idx * HEADS + h] * LOG2E;
  } else {
#pragma unroll
    for (int h = 0; h < HEADS; ++h)
      biasP[(size_t)h * NTOK * NPAD + e] = 0.0f;
  }
}

// ---------------------------------------------------------------------------
// Kernel 1b: maskP[w][q][keyP] = mask[w][q][key]*log2e (pad keys -> -1e30)
// ---------------------------------------------------------------------------
__global__ __launch_bounds__(256) void mask_pad_k(const float* __restrict__ mask,
                                                  float* __restrict__ maskP) {
  int e = blockIdx.x * 256 + threadIdx.x;           // over NW*NTOK*NPAD
  if (e >= NW * NTOK * NPAD) return;
  int key = e % NPAD;
  int qw  = e / NPAD;                               // w*NTOK + q
  maskP[e] = (key < NTOK) ? mask[(size_t)qw * NTOK + key] * LOG2E : -1e30f;
}

// ---------------------------------------------------------------------------
// Kernel 2/4: C[M,Nc] = A[M,K] * W[K,Nc] + bias, bf16 WMMA, tile 128x64.
// 256 threads = 8 waves, each wave a 32x32 accumulator (2x2 wmma tiles).
// Double-buffered LDS: next k-tile's global loads overlap current WMMAs.
// ---------------------------------------------------------------------------
template <typename InT, typename OutT>
__global__ __launch_bounds__(256) void gemm_wmma_k(const InT* __restrict__ A,
                                                   const float* __restrict__ W,
                                                   const float* __restrict__ bias,
                                                   OutT* __restrict__ C,
                                                   int M, int K, int Nc) {
  __shared__ __bf16 As[2][128 * 32];   // [buf][row][k]
  __shared__ __bf16 Ws[2][64 * 32];    // [buf][col][k] (transposed on load)

  const int tid  = threadIdx.x;
  const int lane = tid & 31;
  const int wave = tid >> 5;
  const int waveM = wave & 3;          // 4 waves along M
  const int waveN = wave >> 2;         // 2 waves along N
  const int colw = lane & 15;
  const int kb   = (lane >> 4) * 8;

  const int rowBase = blockIdx.y * 128;
  const int colBase = blockIdx.x * 64;

  auto stage = [&](int k0, int buf) {
    if constexpr (sizeof(InT) == 4) {
#pragma unroll
      for (int e = tid * 4; e < 128 * 32; e += 256 * 4) {
        int r = e >> 5, c = e & 31;
        f32x4 fv = *(const f32x4*)(A + (size_t)(rowBase + r) * K + k0 + c);
#pragma unroll
        for (int i = 0; i < 4; ++i) As[buf][r * 32 + c + i] = (__bf16)fv[i];
      }
    } else {
#pragma unroll
      for (int e = tid * 8; e < 128 * 32; e += 256 * 8) {
        int r = e >> 5, c = e & 31;
        *(bf16x8*)(As[buf] + r * 32 + c) =
            *(const bf16x8*)((const __bf16*)A + (size_t)(rowBase + r) * K + k0 + c);
      }
    }
#pragma unroll
    for (int e = tid; e < 32 * 16; e += 256) {
      int kk = e >> 4, n4 = (e & 15) * 4;
      f32x4 wv = *(const f32x4*)(W + (size_t)(k0 + kk) * Nc + colBase + n4);
#pragma unroll
      for (int i = 0; i < 4; ++i) Ws[buf][(n4 + i) * 32 + kk] = (__bf16)wv[i];
    }
  };

  v8f acc[2][2];
#pragma unroll
  for (int mi = 0; mi < 2; ++mi)
#pragma unroll
    for (int ni = 0; ni < 2; ++ni) acc[mi][ni] = (v8f){};

  const int KT = K >> 5;
  stage(0, 0);
  __syncthreads();

  for (int kt = 0; kt < KT; ++kt) {
    const int cur = kt & 1;
    if (kt + 1 < KT) stage((kt + 1) << 5, cur ^ 1);  // overlap with wmma below

    bf16x16 af[2], bfr[2];
#pragma unroll
    for (int mi = 0; mi < 2; ++mi)
      af[mi] = load_frag(As[cur] + (waveM * 32 + mi * 16 + colw) * 32, kb);
#pragma unroll
    for (int ni = 0; ni < 2; ++ni)
      bfr[ni] = load_frag(Ws[cur] + (waveN * 32 + ni * 16 + colw) * 32, kb);

#pragma unroll
    for (int mi = 0; mi < 2; ++mi)
#pragma unroll
      for (int ni = 0; ni < 2; ++ni)
        acc[mi][ni] = wmma_bf16(af[mi], bfr[ni], acc[mi][ni]);
    __syncthreads();
  }

  // ---- epilogue: + bias, cast, store ----
#pragma unroll
  for (int mi = 0; mi < 2; ++mi) {
#pragma unroll
    for (int ni = 0; ni < 2; ++ni) {
      int col = colBase + waveN * 32 + ni * 16 + colw;
      float bv = bias[col];
#pragma unroll
      for (int i = 0; i < 8; ++i) {
        int row = rowBase + waveM * 32 + mi * 16 + i + 8 * (lane >> 4);
        C[(size_t)row * Nc + col] = (OutT)(acc[mi][ni][i] + bv);
      }
    }
  }
}

// ---------------------------------------------------------------------------
// Kernel 3: transposed flash attention, one block per (window, head).
// S^T tiles: lane = query column -> lane-local softmax stats and the P^T
// C-tile registers feed the O^T = V^T * P^T wmma B operand with zero shuffles.
// ---------------------------------------------------------------------------
__global__ __launch_bounds__(256) void attn_wmma_k(const __bf16* __restrict__ qkv,
                                                   const float* __restrict__ biasP,
                                                   const float* __restrict__ maskP,
                                                   __bf16* __restrict__ outp) {
  __shared__ __bf16 ks[NPAD * 32];     // keys [n][d]
  __shared__ __bf16 vs[32 * NPAD];     // values transposed [d][n]

  const int bh = blockIdx.x;
  const int b  = bh >> 3;
  const int h  = bh & 7;
  const int tid  = threadIdx.x;
  const int lane = tid & 31;
  const int wave = tid >> 5;
  const int colw = lane & 15;
  const int half = lane >> 4;
  const int kb   = half * 8;
  const float sscale = 0.17677669529663687f * LOG2E;  // 1/sqrt(32) * log2(e)

  // zero only the 9 pad rows/cols (disjoint from the payload fills below)
  for (int i = tid; i < (NPAD - NTOK) * 32; i += 256) ks[NTOK * 32 + i] = (__bf16)0.0f;
  for (int i = tid; i < 32 * (NPAD - NTOK); i += 256) {
    int d = i / (NPAD - NTOK), n = i % (NPAD - NTOK);
    vs[d * NPAD + NTOK + n] = (__bf16)0.0f;
  }

  const __bf16* base = qkv + (size_t)b * NTOK * 768 + h * HD;
  for (int e = tid; e < NTOK * 4; e += 256) {
    int n = e >> 2, dc = (e & 3) * 8;
    bf16x8 kv = *(const bf16x8*)(base + (size_t)n * 768 + 256 + dc);
    *(bf16x8*)(ks + n * 32 + dc) = kv;
    bf16x8 vv = *(const bf16x8*)(base + (size_t)n * 768 + 512 + dc);
#pragma unroll
    for (int i = 0; i < 8; ++i) vs[(dc + i) * NPAD + n] = vv[i];
  }
  __syncthreads();

  const float* biash = biasP + (size_t)h * NTOK * NPAD;
  const float* maskh = maskP + (size_t)(b & (NW - 1)) * NTOK * NPAD;

  for (int rt = wave; rt < NT; rt += 8) {
    const int q  = rt * 16 + colw;                  // this lane's query
    const int qc = q > NTOK - 1 ? NTOK - 1 : q;     // clamp pad queries
    // B operand Q^T: lane = query column, k = hd (loop-invariant per row tile)
    bf16x16 bq = load_frag(base + (size_t)qc * 768, kb);
    const float* brow = biash + (size_t)qc * NPAD;
    const float* mrow = maskh + (size_t)qc * NPAD;

    v8f o0 = (v8f){}, o1 = (v8f){};   // O^T tiles: hd rows 0..15 / 16..31
    float m = -1e30f, l = 0.0f;       // lane-local stats for this query

    for (int jp = 0; jp < NT / 2; ++jp) {
      const int c0 = jp * 2;
      // ---- S^T = K * Q^T : A = key rows from LDS ----
      bf16x16 ak0 = load_frag(ks + (c0 * 16 + colw) * 32, kb);
      bf16x16 ak1 = load_frag(ks + ((c0 + 1) * 16 + colw) * 32, kb);
      v8f z = (v8f){};
      v8f s0 = wmma_bf16(ak0, bq, z);  // keys c0*16 + i + 8*half
      v8f s1 = wmma_bf16(ak1, bq, z);  // keys (c0+1)*16 + i + 8*half

      // bias+mask: contiguous 8-float runs over the key dim (log2e pre-folded)
      const int kbase = c0 * 16 + 8 * half;
      f32x4 bb[4], mm[4];
      bb[0] = *(const f32x4*)(brow + kbase);      bb[1] = *(const f32x4*)(brow + kbase + 4);
      bb[2] = *(const f32x4*)(brow + kbase + 16); bb[3] = *(const f32x4*)(brow + kbase + 20);
      mm[0] = *(const f32x4*)(mrow + kbase);      mm[1] = *(const f32x4*)(mrow + kbase + 4);
      mm[2] = *(const f32x4*)(mrow + kbase + 16); mm[3] = *(const f32x4*)(mrow + kbase + 20);

      float ev[16];
      float mx = -1e30f;
#pragma unroll
      for (int i = 0; i < 8; ++i) {
        float sv = s0[i] * sscale + bb[i >> 2][i & 3] + mm[i >> 2][i & 3];
        ev[i] = sv; mx = fmaxf(mx, sv);
      }
#pragma unroll
      for (int i = 0; i < 8; ++i) {
        float sv = s1[i] * sscale + bb[2 + (i >> 2)][i & 3] + mm[2 + (i >> 2)][i & 3];
        ev[8 + i] = sv; mx = fmaxf(mx, sv);
      }
      // merge partner half-lane (other 16 keys of this 32-key chunk)
      mx = fmaxf(mx, __shfl_xor(mx, 16, 32));
      float mn = fmaxf(m, mx);
      float alpha = __builtin_amdgcn_exp2f(m - mn);   // raw v_exp_f32
      float rs = 0.0f;
#pragma unroll
      for (int i = 0; i < 16; ++i) {
        ev[i] = __builtin_amdgcn_exp2f(ev[i] - mn);   // no libm range fixups
        rs += ev[i];
      }
      rs += __shfl_xor(rs, 16, 32);
      l = l * alpha + rs;
      m = mn;
#pragma unroll
      for (int i = 0; i < 8; ++i) { o0[i] *= alpha; o1[i] *= alpha; }

      // ---- P^T registers ARE the B fragment for O^T = V^T * P^T ----
      bf16x16 pb;
#pragma unroll
      for (int i = 0; i < 16; ++i) pb[i] = (__bf16)ev[i];
      bf16x16 av0 = load_frag(vs + colw * NPAD + c0 * 16, kb);        // hd 0..15
      bf16x16 av1 = load_frag(vs + (16 + colw) * NPAD + c0 * 16, kb); // hd 16..31
      o0 = wmma_bf16(av0, pb, o0);
      o1 = wmma_bf16(av1, pb, o1);
    }

    // ---- normalize, store two contiguous bf16x8 runs per lane ----
    if (q < NTOK) {
      float inv = 1.0f / l;
      __bf16* op = outp + ((size_t)b * NTOK + q) * CDIM + h * HD;
      bf16x8 r0, r1;
#pragma unroll
      for (int i = 0; i < 8; ++i) {
        r0[i] = (__bf16)(o0[i] * inv);   // hd = 8*half + i
        r1[i] = (__bf16)(o1[i] * inv);   // hd = 16 + 8*half + i
      }
      *(bf16x8*)(op + 8 * half) = r0;
      *(bf16x8*)(op + 16 + 8 * half) = r1;
    }
  }
}

// ---------------------------------------------------------------------------
// Launch
// ---------------------------------------------------------------------------
extern "C" void kernel_launch(void* const* d_in, const int* in_sizes, int n_in,
                              void* d_out, int out_size, void* d_ws, size_t ws_size,
                              hipStream_t stream) {
  const float* x      = (const float*)d_in[0];
  const float* mask   = (const float*)d_in[1];
  const float* qkv_w  = (const float*)d_in[2];
  const float* qkv_b  = (const float*)d_in[3];
  const float* proj_w = (const float*)d_in[4];
  const float* proj_b = (const float*)d_in[5];
  const float* rpb    = (const float*)d_in[6];
  const int*   ridx   = (const int*)d_in[7];
  float* out = (float*)d_out;

  const int M = BW * NTOK;  // 87808
  char* ws = (char*)d_ws;
  __bf16* qkv_ws  = (__bf16*)ws;                                   // M*768 bf16
  size_t  off1    = (size_t)M * 768 * sizeof(__bf16);
  __bf16* attn_ws = (__bf16*)(ws + off1);                          // M*256 bf16
  size_t  off2    = off1 + (size_t)M * CDIM * sizeof(__bf16);
  float*  biasP   = (float*)(ws + off2);                           // 8*343*352 f32
  size_t  off3    = off2 + (size_t)HEADS * NTOK * NPAD * sizeof(float);
  float*  maskP   = (float*)(ws + off3);                           // 64*343*352 f32

  bias_pad_k<<<(NTOK * NPAD + 255) / 256, 256, 0, stream>>>(rpb, ridx, biasP);
  mask_pad_k<<<(NW * NTOK * NPAD + 255) / 256, 256, 0, stream>>>(mask, maskP);

  gemm_wmma_k<float, __bf16><<<dim3(768 / 64, M / 128), 256, 0, stream>>>(
      x, qkv_w, qkv_b, qkv_ws, M, CDIM, 768);

  attn_wmma_k<<<BW * HEADS, 256, 0, stream>>>(qkv_ws, biasP, maskP, attn_ws);

  gemm_wmma_k<__bf16, float><<<dim3(CDIM / 64, M / 128), 256, 0, stream>>>(
      attn_ws, proj_w, proj_b, out, M, CDIM, CDIM);
}
